// Pct_46832323395811
// MI455X (gfx1250) — matrix operand order repaired and checked by reading, not compiled, not run on hardware
//
#include <hip/hip_runtime.h>
#include <hip/hip_bf16.h>
#include <stdint.h>

typedef __attribute__((ext_vector_type(16))) _Float16 v16h;
typedef __attribute__((ext_vector_type(8)))  _Float16 v8h;
typedef __attribute__((ext_vector_type(8)))  float    v8f;

#define BQ 4
#define NQ 1024
#define NSAMP 32
#define CEILDIV(a,b) (((a)+(b)-1)/(b))

// ---------------------------------------------------------------------------
// WMMA GEMM:  C[bz] = A[bz](MxK) * B[bz](KxNcols) (+bias per-row, compile-time)
// AT=1: A stored transposed (K x M)  (used for the k^T k attention energy).
// blockDim = 256 (8 waves). Block computes a 64 x 128 tile of C:
//   wave w owns columns n0 + w*16; 4 M-subtiles accumulated per wave
//   (B fragment loaded once per k-step, reused by 4 WMMAs -> 4x less B traffic).
// LDS staging with f32->f16 conversion:
//   As[64][32]  (row-major, 64B rows)         -> A fragments: 2x ds_load_b128
//   Bs[128][40] (transposed [n][k], padded)   -> B fragments: 2x ds_load_b128
// K multiple of 32, M multiple of 64, Ncols multiple of 128.
// ---------------------------------------------------------------------------
template<int AT, int HASB>
__global__ __launch_bounds__(256)
void k_wmma_gemm(const float* __restrict__ A, long long aBatch,
                 const float* __restrict__ Bm, long long bBatch,
                 float* __restrict__ C, long long cBatch,
                 const float* __restrict__ bias,
                 int M, int K, int Ncols)
{
    __shared__ _Float16 As[64 * 32];        // 4 KB
    __shared__ _Float16 Bs[128 * 40];       // 10 KB (stride 40 halves = 80B, 16B aligned)

    const int tid    = threadIdx.x;
    const int lane   = tid & 31;
    const int wave   = tid >> 5;
    const int laneHi = lane >> 4;
    const int lane15 = lane & 15;
    const int tm0    = blockIdx.y * 64;
    const int n0     = blockIdx.x * 128;
    const int bz     = blockIdx.z;

    const float* Ab = A  + (size_t)bz * aBatch;
    const float* Bb = Bm + (size_t)bz * bBatch;
    float*       Cb = C  + (size_t)bz * cBatch;

    // staging assignments (fixed per thread)
    const int sa_m = tid >> 2;              // A: 8 consecutive k per thread, rows 0..63
    const int sa_k = (tid & 3) * 8;
    const int sb_n = tid & 127;             // B: 16 consecutive k per thread
    const int sb_k = (tid >> 7) << 4;

    const int nloc = wave * 16 + lane15;    // fragment column within block

    v8f acc[4] = {};
    for (int kk = 0; kk < K; kk += 32) {
        __syncthreads();
        // ---- stage A tile (64x32) ----
        {
            v8h h;
            if (AT) {
#pragma unroll
                for (int j = 0; j < 8; ++j)
                    h[j] = (_Float16)Ab[(size_t)(kk + sa_k + j) * M + tm0 + sa_m];
            } else {
                const float4 f0 = *(const float4*)(Ab + (size_t)(tm0 + sa_m) * K + kk + sa_k);
                const float4 f1 = *(const float4*)(Ab + (size_t)(tm0 + sa_m) * K + kk + sa_k + 4);
                h[0] = (_Float16)f0.x; h[1] = (_Float16)f0.y;
                h[2] = (_Float16)f0.z; h[3] = (_Float16)f0.w;
                h[4] = (_Float16)f1.x; h[5] = (_Float16)f1.y;
                h[6] = (_Float16)f1.z; h[7] = (_Float16)f1.w;
            }
            *(v8h*)(As + sa_m * 32 + sa_k) = h;
        }
        // ---- stage B tile (32x128, transposed into [n][k]) ----
        {
            const float* bp = Bb + (size_t)(kk + sb_k) * Ncols + n0 + sb_n;
            v8h lo, hi;
#pragma unroll
            for (int j = 0; j < 8; ++j) {
                lo[j] = (_Float16)bp[(size_t)j * Ncols];
                hi[j] = (_Float16)bp[(size_t)(j + 8) * Ncols];
            }
            *(v8h*)(Bs + sb_n * 40 + sb_k)     = lo;
            *(v8h*)(Bs + sb_n * 40 + sb_k + 8) = hi;
        }
        __syncthreads();
        // ---- B fragment (loaded once, reused by 4 WMMAs) ----
        const v8h blo = *(const v8h*)(Bs + nloc * 40 + laneHi * 8);
        const v8h bhi = *(const v8h*)(Bs + nloc * 40 + laneHi * 8 + 16);
        v16h bf;
#pragma unroll
        for (int j = 0; j < 8; ++j) { bf[j] = blo[j]; bf[j + 8] = bhi[j]; }
        // ---- 4 M-subtiles ----
#pragma unroll
        for (int mi = 0; mi < 4; ++mi) {
            const v8h alo = *(const v8h*)(As + (mi * 16 + lane15) * 32 + laneHi * 8);
            const v8h ahi = *(const v8h*)(As + (mi * 16 + lane15) * 32 + laneHi * 8 + 16);
            v16h af;
#pragma unroll
            for (int j = 0; j < 8; ++j) { af[j] = alo[j]; af[j + 8] = ahi[j]; }
            acc[mi] = __builtin_amdgcn_wmma_f32_16x16x32_f16(false, af, false, bf,
                                                             (short)0, acc[mi], false, false);
        }
    }

    const int gn = n0 + nloc;
#pragma unroll
    for (int mi = 0; mi < 4; ++mi) {
#pragma unroll
        for (int r = 0; r < 8; ++r) {
            const int mrow = tm0 + mi * 16 + r + laneHi * 8;
            float v = acc[mi][r];
            if (HASB) v += bias[mrow];
            Cb[(size_t)mrow * Ncols + gn] = v;
        }
    }
}

// ---------------------------------------------------------------------------
// Small-K pointwise conv (K=6 / K=3): Y[b,o,n] = sum_c W[o,c]*X[b,c,n] (+bias)
// X element addressed as X[b*xb + c*xc + n*xn]
// ---------------------------------------------------------------------------
__global__ void k_gemm_small(const float* __restrict__ W, const float* __restrict__ X,
                             float* __restrict__ Y, const float* __restrict__ bias,
                             int M, int K, int Nn,
                             long long xb, long long xc, long long xn, long long total)
{
    long long i = (long long)blockIdx.x * blockDim.x + threadIdx.x;
    if (i >= total) return;
    int n = (int)(i % Nn);
    int o = (int)((i / Nn) % M);
    int b = (int)(i / ((long long)Nn * M));
    float s = bias ? bias[o] : 0.f;
    for (int c = 0; c < K; ++c)
        s += W[o * K + c] * X[(long long)b * xb + (long long)c * xc + (long long)n * xn];
    Y[i] = s;
}

// Per-channel mean/var (population) over Bt batches x M cols. X[b*bStride + c*M + m]
__global__ void k_rowstat(const float* __restrict__ X, float* __restrict__ mean,
                          float* __restrict__ var, int C, long long M,
                          int Bt, long long bStride)
{
    int c = blockIdx.x;
    int t = threadIdx.x;
    __shared__ float s1[256], s2[256];
    float a1 = 0.f, a2 = 0.f;
    for (int b = 0; b < Bt; ++b) {
        const float* p = X + (long long)b * bStride + (long long)c * M;
        for (long long m = t; m < M; m += 256) { float v = p[m]; a1 += v; a2 += v * v; }
    }
    s1[t] = a1; s2[t] = a2; __syncthreads();
    for (int s = 128; s > 0; s >>= 1) {
        if (t < s) { s1[t] += s1[t + s]; s2[t] += s2[t + s]; }
        __syncthreads();
    }
    if (t == 0) {
        float cnt = (float)((long long)Bt * M);
        float mu = s1[0] / cnt;
        mean[c] = mu;
        var[c]  = s2[0] / cnt - mu * mu;
    }
}

// y = g*(x-mean)*rsqrt(var+eps)+b, act: 0 none, 1 relu, 2 lrelu(0.2)
__global__ void k_bn_act(float* __restrict__ Y, const float* __restrict__ X,
                         const float* __restrict__ mean, const float* __restrict__ var,
                         const float* __restrict__ g, const float* __restrict__ bb,
                         int C, long long M, long long total, int act)
{
    long long i = (long long)blockIdx.x * blockDim.x + threadIdx.x;
    if (i >= total) return;
    int c = (int)((i / M) % C);
    float v = (X[i] - mean[c]) * rsqrtf(var[c] + 1e-5f) * g[c] + bb[c];
    if (act == 1)      v = fmaxf(v, 0.f);
    else if (act == 2) v = (v >= 0.f) ? v : 0.2f * v;
    Y[i] = v;
}

// (B,3,N) channel-major -> (B,N,3) point-major
__global__ void k_to_pm(float* __restrict__ dst, const float* __restrict__ src, long long total)
{
    long long i = (long long)blockIdx.x * blockDim.x + threadIdx.x;
    if (i >= total) return;
    int d = (int)(i % 3);
    int n = (int)((i / 3) % NQ);
    int b = (int)(i / (3LL * NQ));
    dst[i] = src[(long long)b * 3 * NQ + (long long)d * NQ + n];
}

// Farthest point sampling: one block per batch; matches jnp.argmax (first max).
__global__ void k_fps(const float* __restrict__ xyz, int* __restrict__ out, int Nn)
{
    const int b = blockIdx.x;
    const int t = threadIdx.x;
    __shared__ float sdist[NQ];
    __shared__ float rmax[256];
    __shared__ int   rarg[256];
    __shared__ int   sfar;
    const float* base = xyz + (size_t)b * Nn * 3;
    for (int i = t; i < Nn; i += 256) sdist[i] = 1e10f;
    if (t == 0) sfar = 0;
    __syncthreads();
    for (int it = 0; it < Nn; ++it) {
        int far = sfar;
        if (t == 0) out[b * Nn + it] = far;
        float cx = base[far * 3], cy = base[far * 3 + 1], cz = base[far * 3 + 2];
        float lm = -1.f; int la = 0x7fffffff;
        for (int i = t; i < Nn; i += 256) {
            float dx = base[i * 3] - cx, dy = base[i * 3 + 1] - cy, dz = base[i * 3 + 2] - cz;
            float d = dx * dx + dy * dy + dz * dz;
            float nd = fminf(sdist[i], d);
            sdist[i] = nd;
            if (nd > lm) { lm = nd; la = i; }
        }
        rmax[t] = lm; rarg[t] = la;
        __syncthreads();
        for (int s = 128; s > 0; s >>= 1) {
            if (t < s) {
                if (rmax[t + s] > rmax[t] ||
                    (rmax[t + s] == rmax[t] && rarg[t + s] < rarg[t])) {
                    rmax[t] = rmax[t + s]; rarg[t] = rarg[t + s];
                }
            }
            __syncthreads();
        }
        if (t == 0) sfar = rarg[0];
        __syncthreads();
    }
}

// new_xyz[b,s,:] = xyz[b, fps[b,s], :]
__global__ void k_gather_xyz(float* __restrict__ dst, const float* __restrict__ src,
                             const int* __restrict__ fps, long long total)
{
    long long i = (long long)blockIdx.x * blockDim.x + threadIdx.x;
    if (i >= total) return;
    int d = (int)(i % 3);
    int s = (int)((i / 3) % NQ);
    int b = (int)(i / (3LL * NQ));
    int idx = fps[b * NQ + s];
    dst[i] = src[((long long)b * NQ + idx) * 3 + d];
}

// query_ball: first NSAMP indices (ascending) with dist^2<=r^2, padded with first.
__global__ void k_qball(const float* __restrict__ xyz, const float* __restrict__ ctr,
                        int* __restrict__ out, int Nn, float r2, long long total)
{
    long long g = (long long)blockIdx.x * blockDim.x + threadIdx.x;
    if (g >= total) return;
    int b = (int)(g / NQ);
    const float* pb = xyz + (size_t)b * Nn * 3;
    float cx = ctr[g * 3], cy = ctr[g * 3 + 1], cz = ctr[g * 3 + 2];
    int* o = out + g * NSAMP;
    int cnt = 0, first = 0;
    for (int n = 0; n < Nn && cnt < NSAMP; ++n) {
        float dx = pb[n * 3] - cx, dy = pb[n * 3 + 1] - cy, dz = pb[n * 3 + 2] - cz;
        if (dx * dx + dy * dy + dz * dz <= r2) {
            if (cnt == 0) first = n;
            o[cnt++] = n;
        }
    }
    for (int j = cnt; j < NSAMP; ++j) o[j] = first;
}

// Xg[c, (b*N+s)*32+j] = c<D ? pts[b,c,idx]-pts[b,c,ctr] : pts[b,c-D,ctr]
__global__ void k_group(const float* __restrict__ pts, const int* __restrict__ ball,
                        const int* __restrict__ fps, float* __restrict__ Xg,
                        int D, int Nn, long long Mcols, long long total)
{
    long long i = (long long)blockIdx.x * blockDim.x + threadIdx.x;
    if (i >= total) return;
    int c = (int)(i / Mcols);
    long long m = i % Mcols;
    int b = (int)(m / ((long long)NQ * NSAMP));
    long long rem = m % ((long long)NQ * NSAMP);
    int s = (int)(rem / NSAMP);
    int j = (int)(rem % NSAMP);
    int ci = fps[b * NQ + s];
    float center = pts[(long long)b * D * Nn + (long long)(c % D) * Nn + ci];
    float val;
    if (c < D) {
        int ni = ball[((long long)b * NQ + s) * NSAMP + j];
        val = pts[(long long)b * D * Nn + (long long)c * Nn + ni] - center;
    } else {
        val = center;
    }
    Xg[i] = val;
}

// pooled[b,o,n] = max_j H[o, (b*N+n)*32+j]
__global__ void k_maxpool(const float* __restrict__ H, float* __restrict__ out,
                          int O, long long total)
{
    long long i = (long long)blockIdx.x * blockDim.x + threadIdx.x;
    if (i >= total) return;
    int n = (int)(i % NQ);
    int o = (int)((i / NQ) % O);
    int b = (int)(i / ((long long)NQ * O));
    long long m0 = ((long long)b * NQ + n) * NSAMP;
    const float* p = H + (long long)o * (BQ * (long long)NQ * NSAMP) + m0;
    float mx = -3.4e38f;
    for (int j = 0; j < NSAMP; ++j) mx = fmaxf(mx, p[j]);
    out[i] = mx;
}

// Row softmax over last dim (contiguous), one block per row.
__global__ void k_softmax(float* __restrict__ E, int Mcols)
{
    long long base = (long long)blockIdx.x * Mcols;
    int t = threadIdx.x;
    __shared__ float red[256];
    float mx = -3.4e38f;
    for (int m = t; m < Mcols; m += 256) mx = fmaxf(mx, E[base + m]);
    red[t] = mx; __syncthreads();
    for (int s = 128; s > 0; s >>= 1) { if (t < s) red[t] = fmaxf(red[t], red[t + s]); __syncthreads(); }
    mx = red[0]; __syncthreads();
    float sum = 0.f;
    for (int m = t; m < Mcols; m += 256) { float e = expf(E[base + m] - mx); E[base + m] = e; sum += e; }
    red[t] = sum; __syncthreads();
    for (int s = 128; s > 0; s >>= 1) { if (t < s) red[t] += red[t + s]; __syncthreads(); }
    float inv = 1.f / red[0];
    for (int m = t; m < Mcols; m += 256) E[base + m] *= inv;
}

// colsum[b,m] = sum_n att[b,n,m]
__global__ void k_colsum(const float* __restrict__ att, float* __restrict__ cs, long long total)
{
    long long i = (long long)blockIdx.x * blockDim.x + threadIdx.x;
    if (i >= total) return;
    int m = (int)(i % NQ);
    int b = (int)(i / NQ);
    const float* p = att + (long long)b * NQ * NQ + m;
    float s = 0.f;
    for (int n = 0; n < NQ; ++n) s += p[(long long)n * NQ];
    cs[i] = s;
}

__global__ void k_attnorm(float* __restrict__ att, const float* __restrict__ cs, long long total)
{
    long long i = (long long)blockIdx.x * blockDim.x + threadIdx.x;
    if (i >= total) return;
    int m = (int)(i % NQ);
    int b = (int)(i / ((long long)NQ * NQ));
    att[i] /= (1e-9f + cs[(long long)b * NQ + m]);
}

__global__ void k_add(float* __restrict__ o, const float* __restrict__ a,
                      const float* __restrict__ b, long long n)
{
    long long i = (long long)blockIdx.x * blockDim.x + threadIdx.x;
    if (i < n) o[i] = a[i] + b[i];
}
__global__ void k_sub(float* __restrict__ o, const float* __restrict__ a,
                      const float* __restrict__ b, long long n)
{
    long long i = (long long)blockIdx.x * blockDim.x + threadIdx.x;
    if (i < n) o[i] = a[i] - b[i];
}

// dst[b, cOff+c, n] = src[b, c, n]
__global__ void k_copy_chs(float* __restrict__ dst, const float* __restrict__ src,
                           int C, int Nn, int dstC, int cOff, long long total)
{
    long long i = (long long)blockIdx.x * blockDim.x + threadIdx.x;
    if (i >= total) return;
    long long r = i % ((long long)C * Nn);
    int b = (int)(i / ((long long)C * Nn));
    dst[(long long)b * dstC * Nn + (long long)cOff * Nn + r] = src[i];
}

// ---------------------------------------------------------------------------
// Host-side orchestration
// ---------------------------------------------------------------------------
extern "C" void kernel_launch(void* const* d_in, const int* in_sizes, int n_in,
                              void* d_out, int out_size, void* d_ws, size_t ws_size,
                              hipStream_t stream)
{
    enum {
        IN_X = 0, IN_FMAP,
        P_CONV1_W, P_BN1_G, P_BN1_B, P_CONV2_W, P_BN2_G, P_BN2_B,
        P_L0_C1, P_L0_C2, P_L0_BN1_G, P_L0_BN1_B, P_L0_BN2_G, P_L0_BN2_B,
        P_L1_C1, P_L1_C2, P_L1_BN1_G, P_L1_BN1_B, P_L1_BN2_G, P_L1_BN2_B,
        P_PT_C1, P_PT_BN1_G, P_PT_BN1_B, P_POS_W, P_POS_B,
        P_SA1_Q, P_SA1_V, P_SA1_VB, P_SA1_T, P_SA1_TB, P_SA1_BNG, P_SA1_BNB,
        P_SA2_Q, P_SA2_V, P_SA2_VB, P_SA2_T, P_SA2_TB, P_SA2_BNG, P_SA2_BNB,
        P_SA3_Q, P_SA3_V, P_SA3_VB, P_SA3_T, P_SA3_TB, P_SA3_BNG, P_SA3_BNB,
        P_F1_W, P_F1_G, P_F1_B, P_F2_W, P_F2_G, P_F2_B, P_F3_W, P_F3_G, P_F3_B
    };
    auto IN = [&](int i) -> const float* { return (const float*)d_in[i]; };

    uintptr_t wbase = (uintptr_t)d_ws;
    size_t off = 0;
    auto alloc = [&](size_t bytes) -> void* {
        off = (off + 255) & ~(size_t)255;
        void* p = (void*)(wbase + off);
        off += bytes;
        return p;
    };
    const long long GM = (long long)BQ * NQ * NSAMP;   // 131072 grouped columns

    float* h0    = (float*)alloc((size_t)BQ * 6 * NQ * 4);
    float* h1    = (float*)alloc((size_t)BQ * 64 * NQ * 4);
    float* h2    = (float*)alloc((size_t)BQ * 64 * NQ * 4);
    float* meanb = (float*)alloc(1024 * 4);
    float* varb  = (float*)alloc(1024 * 4);
    float* xyz1  = (float*)alloc((size_t)BQ * NQ * 3 * 4);
    int*   fpsi  = (int*)  alloc((size_t)BQ * NQ * 4);
    float* nxyz1 = (float*)alloc((size_t)BQ * NQ * 3 * 4);
    float* nxyz2 = (float*)alloc((size_t)BQ * NQ * 3 * 4);
    int*   ball  = (int*)  alloc((size_t)BQ * NQ * NSAMP * 4);
    float* XgB   = (float*)alloc((size_t)256 * GM * 4);   // Xg, reused as 2nd GEMM output
    float* Ha    = (float*)alloc((size_t)256 * GM * 4);
    float* f0    = (float*)alloc((size_t)BQ * 128 * NQ * 4);
    float* f1f   = (float*)alloc((size_t)BQ * 256 * NQ * 4);
    float* xyzf  = (float*)alloc((size_t)BQ * 256 * NQ * 4);
    float* x0b   = (float*)alloc((size_t)BQ * 256 * NQ * 4);
    float* x1b   = (float*)alloc((size_t)BQ * 256 * NQ * 4);
    float* x2b   = (float*)alloc((size_t)BQ * 256 * NQ * 4);
    float* x3b   = (float*)alloc((size_t)BQ * 256 * NQ * 4);
    float* xa    = (float*)alloc((size_t)BQ * 256 * NQ * 4);
    float* kf    = (float*)alloc((size_t)BQ * 64 * NQ * 4);
    float* vf    = (float*)alloc((size_t)BQ * 256 * NQ * 4);
    float* att   = (float*)alloc((size_t)BQ * NQ * NQ * 4);
    float* cs    = (float*)alloc((size_t)BQ * NQ * 4);
    float* xr    = (float*)alloc((size_t)BQ * 256 * NQ * 4);
    float* diff  = (float*)alloc((size_t)BQ * 256 * NQ * 4);
    float* tc    = (float*)alloc((size_t)BQ * 256 * NQ * 4);
    float* cat   = (float*)alloc((size_t)BQ * 1024 * NQ * 4);
    float* g1    = (float*)alloc((size_t)BQ * 512 * NQ * 4);
    float* g2    = (float*)alloc((size_t)BQ * 256 * NQ * 4);
    float* outp  = (float*)d_out;

    const long long E256 = (long long)BQ * 256 * NQ;

    // ---- input MLP: concat -> conv1(K=6)+BN+ReLU -> conv2(K=64 WMMA)+BN+ReLU
    { long long t = (long long)BQ * 3 * NQ;
      k_copy_chs<<<CEILDIV(t,256),256,0,stream>>>(h0, IN(IN_FMAP), 3, NQ, 6, 0, t);
      k_copy_chs<<<CEILDIV(t,256),256,0,stream>>>(h0, IN(IN_X),    3, NQ, 6, 3, t); }
    { long long t = (long long)BQ * 64 * NQ;
      k_gemm_small<<<CEILDIV(t,256),256,0,stream>>>(IN(P_CONV1_W), h0, h1, nullptr,
          64, 6, NQ, 6LL*NQ, NQ, 1, t);
      k_rowstat<<<64,256,0,stream>>>(h1, meanb, varb, 64, NQ, BQ, 64LL*NQ);
      k_bn_act<<<CEILDIV(t,256),256,0,stream>>>(h1, h1, meanb, varb, IN(P_BN1_G), IN(P_BN1_B),
          64, NQ, t, 1);
      k_wmma_gemm<0,0><<<dim3(NQ/128, 1, BQ),256,0,stream>>>(IN(P_CONV2_W), 0, h1, 64LL*NQ,
          h2, 64LL*NQ, nullptr, 64, 64, NQ);
      k_rowstat<<<64,256,0,stream>>>(h2, meanb, varb, 64, NQ, BQ, 64LL*NQ);
      k_bn_act<<<CEILDIV(t,256),256,0,stream>>>(h2, h2, meanb, varb, IN(P_BN2_G), IN(P_BN2_B),
          64, NQ, t, 1); }

    // ---- stage 1 sample_and_group + local_op (D=64 -> 128ch, WMMA K=128)
    { long long t3 = (long long)BQ * NQ * 3;
      k_to_pm<<<CEILDIV(t3,256),256,0,stream>>>(xyz1, IN(IN_X), t3);
      k_fps<<<BQ,256,0,stream>>>(xyz1, fpsi, NQ);
      k_gather_xyz<<<CEILDIV(t3,256),256,0,stream>>>(nxyz1, xyz1, fpsi, t3);
      long long tq = (long long)BQ * NQ;
      k_qball<<<CEILDIV(tq,128),128,0,stream>>>(xyz1, nxyz1, ball, NQ, 0.15f*0.15f, tq);
      long long tg = 128LL * GM;
      k_group<<<CEILDIV(tg,256),256,0,stream>>>(h2, ball, fpsi, XgB, 64, NQ, GM, tg);
      k_wmma_gemm<0,0><<<dim3(GM/128, 2, 1),256,0,stream>>>(IN(P_L0_C1), 0, XgB, 0,
          Ha, 0, nullptr, 128, 128, (int)GM);
      k_rowstat<<<128,256,0,stream>>>(Ha, meanb, varb, 128, GM, 1, 0);
      k_bn_act<<<CEILDIV(tg,256),256,0,stream>>>(Ha, Ha, meanb, varb, IN(P_L0_BN1_G), IN(P_L0_BN1_B),
          128, GM, tg, 1);
      k_wmma_gemm<0,0><<<dim3(GM/128, 2, 1),256,0,stream>>>(IN(P_L0_C2), 0, Ha, 0,
          XgB, 0, nullptr, 128, 128, (int)GM);
      k_rowstat<<<128,256,0,stream>>>(XgB, meanb, varb, 128, GM, 1, 0);
      k_bn_act<<<CEILDIV(tg,256),256,0,stream>>>(XgB, XgB, meanb, varb, IN(P_L0_BN2_G), IN(P_L0_BN2_B),
          128, GM, tg, 1);
      long long tp = (long long)BQ * 128 * NQ;
      k_maxpool<<<CEILDIV(tp,256),256,0,stream>>>(XgB, f0, 128, tp); }

    // ---- stage 2 sample_and_group + local_op (D=128 -> 256ch, WMMA K=256)
    { long long t3 = (long long)BQ * NQ * 3;
      k_fps<<<BQ,256,0,stream>>>(nxyz1, fpsi, NQ);
      k_gather_xyz<<<CEILDIV(t3,256),256,0,stream>>>(nxyz2, nxyz1, fpsi, t3);
      long long tq = (long long)BQ * NQ;
      k_qball<<<CEILDIV(tq,128),128,0,stream>>>(nxyz1, nxyz2, ball, NQ, 0.2f*0.2f, tq);
      long long tg = 256LL * GM;
      k_group<<<CEILDIV(tg,256),256,0,stream>>>(f0, ball, fpsi, XgB, 128, NQ, GM, tg);
      k_wmma_gemm<0,0><<<dim3(GM/128, 4, 1),256,0,stream>>>(IN(P_L1_C1), 0, XgB, 0,
          Ha, 0, nullptr, 256, 256, (int)GM);
      k_rowstat<<<256,256,0,stream>>>(Ha, meanb, varb, 256, GM, 1, 0);
      k_bn_act<<<CEILDIV(tg,256),256,0,stream>>>(Ha, Ha, meanb, varb, IN(P_L1_BN1_G), IN(P_L1_BN1_B),
          256, GM, tg, 1);
      k_wmma_gemm<0,0><<<dim3(GM/128, 4, 1),256,0,stream>>>(IN(P_L1_C2), 0, Ha, 0,
          XgB, 0, nullptr, 256, 256, (int)GM);
      k_rowstat<<<256,256,0,stream>>>(XgB, meanb, varb, 256, GM, 1, 0);
      k_bn_act<<<CEILDIV(tg,256),256,0,stream>>>(XgB, XgB, meanb, varb, IN(P_L1_BN2_G), IN(P_L1_BN2_B),
          256, GM, tg, 1);
      k_maxpool<<<CEILDIV(E256,256),256,0,stream>>>(XgB, f1f, 256, E256); }

    // ---- positional features (K=3) and pt_c1 (WMMA 256x256)
    k_gemm_small<<<CEILDIV(E256,256),256,0,stream>>>(IN(P_POS_W), nxyz2, xyzf, IN(P_POS_B),
        256, 3, NQ, 3LL*NQ, 1, 3, E256);
    k_wmma_gemm<0,0><<<dim3(NQ/128, 4, BQ),256,0,stream>>>(IN(P_PT_C1), 0, f1f, 256LL*NQ,
        x0b, 256LL*NQ, nullptr, 256, 256, NQ);
    k_rowstat<<<256,256,0,stream>>>(x0b, meanb, varb, 256, NQ, BQ, 256LL*NQ);
    k_bn_act<<<CEILDIV(E256,256),256,0,stream>>>(x0b, x0b, meanb, varb, IN(P_PT_BN1_G), IN(P_PT_BN1_B),
        256, NQ, E256, 1);

    // ---- three offset-attention layers
    const int saQ[3]  = {P_SA1_Q,  P_SA2_Q,  P_SA3_Q};
    const int saV[3]  = {P_SA1_V,  P_SA2_V,  P_SA3_V};
    const int saVB[3] = {P_SA1_VB, P_SA2_VB, P_SA3_VB};
    const int saT[3]  = {P_SA1_T,  P_SA2_T,  P_SA3_T};
    const int saTB[3] = {P_SA1_TB, P_SA2_TB, P_SA3_TB};
    const int saG[3]  = {P_SA1_BNG, P_SA2_BNG, P_SA3_BNG};
    const int saBB[3] = {P_SA1_BNB, P_SA2_BNB, P_SA3_BNB};
    float* xprev = x0b;
    float* xouts[3] = {x1b, x2b, x3b};
    for (int i = 0; i < 3; ++i) {
        float* xout = xouts[i];
        k_add<<<CEILDIV(E256,256),256,0,stream>>>(xa, xprev, xyzf, E256);
        // k = Wq * xa  (q == k, shared weights, no bias)
        k_wmma_gemm<0,0><<<dim3(NQ/128, 1, BQ),256,0,stream>>>(IN(saQ[i]), 0, xa, 256LL*NQ,
            kf, 64LL*NQ, nullptr, 64, 256, NQ);
        // v = Wv * xa + vb
        k_wmma_gemm<0,1><<<dim3(NQ/128, 4, BQ),256,0,stream>>>(IN(saV[i]), 0, xa, 256LL*NQ,
            vf, 256LL*NQ, IN(saVB[i]), 256, 256, NQ);
        // energy = k^T k  (A transposed mode)
        k_wmma_gemm<1,0><<<dim3(NQ/128, NQ/64, BQ),256,0,stream>>>(kf, 64LL*NQ, kf, 64LL*NQ,
            att, (long long)NQ*NQ, nullptr, NQ, 64, NQ);
        k_softmax<<<BQ*NQ,256,0,stream>>>(att, NQ);
        { long long tc_ = (long long)BQ * NQ;
          k_colsum<<<CEILDIV(tc_,256),256,0,stream>>>(att, cs, tc_);
          long long ta = (long long)BQ * NQ * NQ;
          k_attnorm<<<CEILDIV(ta,256),256,0,stream>>>(att, cs, ta); }
        // x_r = v @ att
        k_wmma_gemm<0,0><<<dim3(NQ/128, 4, BQ),256,0,stream>>>(vf, 256LL*NQ, att, (long long)NQ*NQ,
            xr, 256LL*NQ, nullptr, 256, NQ, NQ);
        k_sub<<<CEILDIV(E256,256),256,0,stream>>>(diff, xa, xr, E256);
        k_wmma_gemm<0,1><<<dim3(NQ/128, 4, BQ),256,0,stream>>>(IN(saT[i]), 0, diff, 256LL*NQ,
            tc, 256LL*NQ, IN(saTB[i]), 256, 256, NQ);
        k_rowstat<<<256,256,0,stream>>>(tc, meanb, varb, 256, NQ, BQ, 256LL*NQ);
        k_bn_act<<<CEILDIV(E256,256),256,0,stream>>>(tc, tc, meanb, varb, IN(saG[i]), IN(saBB[i]),
            256, NQ, E256, 1);
        k_add<<<CEILDIV(E256,256),256,0,stream>>>(xout, xa, tc, E256);
        xprev = xout;
    }

    // ---- concat [x1,x2,x3,f1] -> fused MLP f1/f2/f3 (WMMA, lrelu BN)
    k_copy_chs<<<CEILDIV(E256,256),256,0,stream>>>(cat, x1b, 256, NQ, 1024, 0,   E256);
    k_copy_chs<<<CEILDIV(E256,256),256,0,stream>>>(cat, x2b, 256, NQ, 1024, 256, E256);
    k_copy_chs<<<CEILDIV(E256,256),256,0,stream>>>(cat, x3b, 256, NQ, 1024, 512, E256);
    k_copy_chs<<<CEILDIV(E256,256),256,0,stream>>>(cat, f1f, 256, NQ, 1024, 768, E256);

    { long long t = (long long)BQ * 512 * NQ;
      k_wmma_gemm<0,0><<<dim3(NQ/128, 8, BQ),256,0,stream>>>(IN(P_F1_W), 0, cat, 1024LL*NQ,
          g1, 512LL*NQ, nullptr, 512, 1024, NQ);
      k_rowstat<<<512,256,0,stream>>>(g1, meanb, varb, 512, NQ, BQ, 512LL*NQ);
      k_bn_act<<<CEILDIV(t,256),256,0,stream>>>(g1, g1, meanb, varb, IN(P_F1_G), IN(P_F1_B),
          512, NQ, t, 2); }
    k_wmma_gemm<0,0><<<dim3(NQ/128, 4, BQ),256,0,stream>>>(IN(P_F2_W), 0, g1, 512LL*NQ,
        g2, 256LL*NQ, nullptr, 256, 512, NQ);
    k_rowstat<<<256,256,0,stream>>>(g2, meanb, varb, 256, NQ, BQ, 256LL*NQ);
    k_bn_act<<<CEILDIV(E256,256),256,0,stream>>>(g2, g2, meanb, varb, IN(P_F2_G), IN(P_F2_B),
        256, NQ, E256, 2);
    { long long t = (long long)BQ * 128 * NQ;
      k_wmma_gemm<0,0><<<dim3(NQ/128, 2, BQ),256,0,stream>>>(IN(P_F3_W), 0, g2, 256LL*NQ,
          outp, 128LL*NQ, nullptr, 128, 256, NQ);
      k_rowstat<<<128,256,0,stream>>>(outp, meanb, varb, 128, NQ, BQ, 128LL*NQ);
      k_bn_act<<<CEILDIV(t,256),256,0,stream>>>(outp, outp, meanb, varb, IN(P_F3_G), IN(P_F3_B),
          128, NQ, t, 2); }
    (void)in_sizes; (void)n_in; (void)out_size; (void)ws_size;
}
